// MultiheadAttention_29343216566756
// MI455X (gfx1250) — matrix operand order repaired and checked by reading, not compiled, not run on hardware
//
#include <hip/hip_runtime.h>
#include <hip/hip_bf16.h>

typedef _Float16 v16h  __attribute__((ext_vector_type(16)));
typedef _Float16 half8 __attribute__((ext_vector_type(8)));
typedef _Float16 half4 __attribute__((ext_vector_type(4)));
typedef float    v8f   __attribute__((ext_vector_type(8)));

#define HIDDEN   1024
#define HEADS    16
#define HEAD_DIM 64
#define BATCH    2
#define SEQ      2048
#define LDT      72   // LDS row stride in halves: 144B = 36 words -> conflict-free over 16 rows

// ---- fragment loaders (CDNA5 WMMA 16x16x32 f16 layouts, wave32) ----
// A 16x32: lane l holds row M=l&15; halves K in {8*hi..8*hi+7, 16+8*hi..23+8*hi}
__device__ __forceinline__ v16h load_fragA(const _Float16* p, int hi) {
  half8 l0 = *(const half8*)(p + 8 * hi);
  half8 l1 = *(const half8*)(p + 16 + 8 * hi);
  return __builtin_shufflevector(l0, l1, 0,1,2,3,4,5,6,7,8,9,10,11,12,13,14,15);
}
// B 32x16: lane l holds col N=l&15; halves K in {8*hi..8*hi+7, 16+8*hi..23+8*hi}
__device__ __forceinline__ v16h load_fragB(const _Float16* p, int hi) {
  half8 l0 = *(const half8*)(p + 8 * hi);
  half8 l1 = *(const half8*)(p + 16 + 8 * hi);
  return __builtin_shufflevector(l0, l1, 0,1,2,3,4,5,6,7,8,9,10,11,12,13,14,15);
}

// =====================================================================
// GEMM: Y[M=4096, N=1024] = X[M,K=1024] * W[N,K]^T + bias
// MODE 0: X is fp32, output f16 scattered to [B,H,S,Dh]
// MODE 1: X is f16 ([B,S,HIDDEN]), output fp32 [B,S,HIDDEN]
// Block 128x128, BK=64, 256 threads = 8 waves (4x2), wave tile 32x64.
// Software-pipelined: next K-tile prefetched to VGPRs during WMMA phase.
// =====================================================================
template <int MODE>
__global__ __launch_bounds__(256) void gemm128(
    const float* __restrict__ Xf, const _Float16* __restrict__ Xh,
    const float* __restrict__ W, const float* __restrict__ bias,
    _Float16* __restrict__ outH, float* __restrict__ outF)
{
  __shared__ _Float16 As[128 * LDT];
  __shared__ _Float16 Bs[128 * LDT];

  const int t    = threadIdx.x;
  const int lane = t & 31, lo = lane & 15, hi = lane >> 4;
  const int wave = t >> 5, wm = wave >> 1, wn = wave & 1;
  const int bN = blockIdx.x * 128, bM = blockIdx.y * 128;
  const int srow = t >> 1, sc0 = (t & 1) * 32;   // staging: 2 threads per row

  v8f acc[2][4] = {};

  // ---- prologue: prefetch tile k0=0 into registers ----
  float4 aRegF[8], bRegF[8];
  half8  aRegH[4];
  {
    if (MODE == 0) {
      const float4* ax = (const float4*)(Xf + (size_t)(bM + srow) * 1024 + sc0);
#pragma unroll
      for (int q = 0; q < 8; ++q) aRegF[q] = ax[q];
    } else {
      const half8* ax = (const half8*)(Xh + (size_t)(bM + srow) * 1024 + sc0);
#pragma unroll
      for (int q = 0; q < 4; ++q) aRegH[q] = ax[q];
    }
    const float4* bx = (const float4*)(W + (size_t)(bN + srow) * 1024 + sc0);
#pragma unroll
    for (int q = 0; q < 8; ++q) bRegF[q] = bx[q];
  }

  for (int k0 = 0; k0 < 1024; k0 += 64) {
    __syncthreads();   // all waves finished reading LDS from previous tile
    // ---- store prefetched tile to LDS (convert f32->f16 here) ----
    if (MODE == 0) {
#pragma unroll
      for (int q = 0; q < 8; ++q) {
        float4 v = aRegF[q];
        half4 hv = { (_Float16)v.x, (_Float16)v.y, (_Float16)v.z, (_Float16)v.w };
        *(half4*)(&As[srow * LDT + sc0 + q * 4]) = hv;
      }
    } else {
#pragma unroll
      for (int q = 0; q < 4; ++q)
        *(half8*)(&As[srow * LDT + sc0 + q * 8]) = aRegH[q];
    }
#pragma unroll
    for (int q = 0; q < 8; ++q) {
      float4 v = bRegF[q];
      half4 hv = { (_Float16)v.x, (_Float16)v.y, (_Float16)v.z, (_Float16)v.w };
      *(half4*)(&Bs[srow * LDT + sc0 + q * 4]) = hv;
    }
    // ---- issue prefetch of next tile (wrapped; in flight during compute) ----
    {
      const int kn = (k0 + 64) & 1023;
      if (MODE == 0) {
        const float4* ax = (const float4*)(Xf + (size_t)(bM + srow) * 1024 + kn + sc0);
#pragma unroll
        for (int q = 0; q < 8; ++q) aRegF[q] = ax[q];
      } else {
        const half8* ax = (const half8*)(Xh + (size_t)(bM + srow) * 1024 + kn + sc0);
#pragma unroll
        for (int q = 0; q < 4; ++q) aRegH[q] = ax[q];
      }
      const float4* bx = (const float4*)(W + (size_t)(bN + srow) * 1024 + kn + sc0);
#pragma unroll
      for (int q = 0; q < 8; ++q) bRegF[q] = bx[q];
    }
    __syncthreads();   // staged tile visible

    // ---- WMMA phase ----
#pragma unroll
    for (int ks = 0; ks < 2; ++ks) {
      v16h af[2], bf[4];
#pragma unroll
      for (int tm = 0; tm < 2; ++tm)
        af[tm] = load_fragA(&As[(wm * 32 + tm * 16 + lo) * LDT + ks * 32], hi);
#pragma unroll
      for (int tn = 0; tn < 4; ++tn)
        bf[tn] = load_fragB(&Bs[(wn * 64 + tn * 16 + lo) * LDT + ks * 32], hi);
#pragma unroll
      for (int tm = 0; tm < 2; ++tm)
#pragma unroll
        for (int tn = 0; tn < 4; ++tn)
          acc[tm][tn] = __builtin_amdgcn_wmma_f32_16x16x32_f16(
              false, af[tm], false, bf[tn], (short)0, acc[tm][tn], false, false);
    }
  }

  // ---- epilogue: C layout = VGPR r, lane: M = r + 8*hi, N = lo ----
#pragma unroll
  for (int tm = 0; tm < 2; ++tm) {
#pragma unroll
    for (int tn = 0; tn < 4; ++tn) {
      const int gcol = bN + wn * 64 + tn * 16 + lo;
      const float bv = bias[gcol];
#pragma unroll
      for (int r = 0; r < 8; ++r) {
        const int grow = bM + wm * 32 + tm * 16 + r + 8 * hi;
        const float val = acc[tm][tn][r] + bv;
        if (MODE == 0) {
          const int bb = grow >> 11, ss = grow & 2047;
          const int hh = gcol >> 6, dd = gcol & 63;
          outH[(((size_t)bb * HEADS + hh) * SEQ + ss) * HEAD_DIM + dd] = (_Float16)val;
        } else {
          outF[(size_t)grow * HIDDEN + gcol] = val;
        }
      }
    }
  }
}

// =====================================================================
// Flash attention: one (b,h) per blockIdx.y, 128 q-rows per block,
// 4 waves x 32 q-rows each; stream K/V in chunks of 64 keys.
// K/V chunks software-pipelined through VGPRs (prefetch issued after
// softmax to keep peak VGPR pressure < 256); P round-trip is
// wave-private (ordered with an explicit s_wait_dscnt, no barrier).
// =====================================================================
__global__ __launch_bounds__(128) void attn64(
    const _Float16* __restrict__ Qp, const _Float16* __restrict__ Kp,
    const _Float16* __restrict__ Vp, _Float16* __restrict__ Og)
{
  __shared__ _Float16 Ks[64 * LDT];        // [key][d]
  __shared__ _Float16 Vt[64 * LDT];        // [d][key]  (transposed)
  __shared__ _Float16 Pw[4 * 32 * LDT];    // per-wave P staging [qrow][key]

  const int t = threadIdx.x, lane = t & 31, lo = lane & 15, hi = lane >> 4;
  const int wave = t >> 5;
  const int bh = blockIdx.y;
  const int b = bh >> 4, h = bh & 15;
  const int q0 = blockIdx.x * 128 + wave * 32;

  const _Float16* Qb = Qp + ((size_t)bh * SEQ + q0) * HEAD_DIM;
  const _Float16* Kb = Kp + (size_t)bh * SEQ * HEAD_DIM;
  const _Float16* Vb = Vp + (size_t)bh * SEQ * HEAD_DIM;
  _Float16* Pme = &Pw[wave * 32 * LDT];

  // Q fragments live in registers for the whole kernel (A layout, global
  // loads), with the 1/sqrt(Dh) score scale folded in once (f16-safe).
  v16h qf[2][2];
#pragma unroll
  for (int tm = 0; tm < 2; ++tm)
#pragma unroll
    for (int ks = 0; ks < 2; ++ks) {
      v16h q = load_fragA(Qb + (tm * 16 + lo) * HEAD_DIM + ks * 32, hi);
#pragma unroll
      for (int e = 0; e < 16; ++e) q[e] = q[e] * (_Float16)0.125f;
      qf[tm][ks] = q;
    }

  v8f o[2][4] = {};
  float mrun[2][8], lrun[2][8];
#pragma unroll
  for (int tm = 0; tm < 2; ++tm)
#pragma unroll
    for (int r = 0; r < 8; ++r) { mrun[tm][r] = -1e30f; lrun[tm][r] = 0.0f; }

  const int srow = t >> 1, sc0 = (t & 1) * 32;

  // ---- prologue: prefetch chunk kb=0 into registers ----
  half8 kReg[4], vReg[4];
  {
    const half8* ksrc = (const half8*)(Kb + (size_t)srow * HEAD_DIM + sc0);
    const half8* vsrc = (const half8*)(Vb + (size_t)srow * HEAD_DIM + sc0);
#pragma unroll
    for (int q = 0; q < 4; ++q) { kReg[q] = ksrc[q]; vReg[q] = vsrc[q]; }
  }

  for (int kb = 0; kb < SEQ; kb += 64) {
    __syncthreads();   // all waves done reading Ks/Vt of previous chunk
    // ---- store prefetched K (as-is) and V (transposed) chunks ----
#pragma unroll
    for (int q = 0; q < 4; ++q)
      *(half8*)(&Ks[srow * LDT + sc0 + q * 8]) = kReg[q];
#pragma unroll
    for (int q = 0; q < 4; ++q) {
      half8 v = vReg[q];
#pragma unroll
      for (int e = 0; e < 8; ++e)
        Vt[(sc0 + q * 8 + e) * LDT + srow] = v[e];
    }
    __syncthreads();   // staged chunk visible

    // ---- S = (Q*scale) * K^T  (f32 accum) ----
    v8f s[2][4] = {};
#pragma unroll
    for (int ks = 0; ks < 2; ++ks) {
      v16h bf[4];
#pragma unroll
      for (int tn = 0; tn < 4; ++tn)
        bf[tn] = load_fragB(&Ks[(tn * 16 + lo) * LDT + ks * 32], hi);
#pragma unroll
      for (int tm = 0; tm < 2; ++tm)
#pragma unroll
        for (int tn = 0; tn < 4; ++tn)
          s[tm][tn] = __builtin_amdgcn_wmma_f32_16x16x32_f16(
              false, qf[tm][ks], false, bf[tn], (short)0, s[tm][tn], false, false);
    }

    // ---- online softmax (fp32), P -> LDS as f16 A-matrix source ----
#pragma unroll
    for (int tm = 0; tm < 2; ++tm) {
#pragma unroll
      for (int r = 0; r < 8; ++r) {
        float cm = fmaxf(fmaxf(s[tm][0][r], s[tm][1][r]),
                         fmaxf(s[tm][2][r], s[tm][3][r]));
        cm = fmaxf(cm, __shfl_xor(cm, 1, 32));
        cm = fmaxf(cm, __shfl_xor(cm, 2, 32));
        cm = fmaxf(cm, __shfl_xor(cm, 4, 32));
        cm = fmaxf(cm, __shfl_xor(cm, 8, 32));
        const float mold = mrun[tm][r];
        const float mnew = fmaxf(mold, cm);
        const float alpha = __expf(mold - mnew);
        float psum = 0.0f;
#pragma unroll
        for (int tn = 0; tn < 4; ++tn) {
          const float p = __expf(s[tm][tn][r] - mnew);
          s[tm][tn][r] = p;
          psum += p;
          o[tm][tn][r] *= alpha;
        }
        psum += __shfl_xor(psum, 1, 32);
        psum += __shfl_xor(psum, 2, 32);
        psum += __shfl_xor(psum, 4, 32);
        psum += __shfl_xor(psum, 8, 32);
        lrun[tm][r] = lrun[tm][r] * alpha + psum;
        mrun[tm][r] = mnew;
#pragma unroll
        for (int tn = 0; tn < 4; ++tn)
          Pme[(tm * 16 + r + 8 * hi) * LDT + tn * 16 + lo] = (_Float16)s[tm][tn][r];
      }
    }

    // ---- issue prefetch of next chunk (wrapped); placed here so the
    // 32 prefetch VGPRs do not overlap the S/softmax register peak ----
    {
      const int kn = (kb + 64) & (SEQ - 1);
      const half8* ksrc = (const half8*)(Kb + (size_t)(kn + srow) * HEAD_DIM + sc0);
      const half8* vsrc = (const half8*)(Vb + (size_t)(kn + srow) * HEAD_DIM + sc0);
#pragma unroll
      for (int q = 0; q < 4; ++q) { kReg[q] = ksrc[q]; vReg[q] = vsrc[q]; }
    }

    // P is wave-private: LDS ops from this wave are in-order; make the
    // stores land before the fragment loads issue (no block barrier needed).
    asm volatile("s_wait_dscnt 0x0" ::: "memory");

    // ---- O += P * V ----
#pragma unroll
    for (int kk = 0; kk < 2; ++kk) {
      v16h pa[2], vb2[4];
#pragma unroll
      for (int tm = 0; tm < 2; ++tm)
        pa[tm] = load_fragA(&Pme[(tm * 16 + lo) * LDT + kk * 32], hi);
#pragma unroll
      for (int tn = 0; tn < 4; ++tn)
        vb2[tn] = load_fragB(&Vt[(tn * 16 + lo) * LDT + kk * 32], hi);
#pragma unroll
      for (int tm = 0; tm < 2; ++tm)
#pragma unroll
        for (int tn = 0; tn < 4; ++tn)
          o[tm][tn] = __builtin_amdgcn_wmma_f32_16x16x32_f16(
              false, pa[tm], false, vb2[tn], (short)0, o[tm][tn], false, false);
    }
  }

  // ---- epilogue: normalize, write [B,S,HIDDEN] f16 ----
#pragma unroll
  for (int tm = 0; tm < 2; ++tm) {
#pragma unroll
    for (int r = 0; r < 8; ++r) {
      const float rinv = 1.0f / lrun[tm][r];
      const int ss = q0 + tm * 16 + r + 8 * hi;
#pragma unroll
      for (int tn = 0; tn < 4; ++tn) {
        const int col = h * 64 + tn * 16 + lo;
        Og[((size_t)b * SEQ + ss) * HIDDEN + col] = (_Float16)(o[tm][tn][r] * rinv);
      }
    }
  }
}

// =====================================================================
extern "C" void kernel_launch(void* const* d_in, const int* in_sizes, int n_in,
                              void* d_out, int out_size, void* d_ws, size_t ws_size,
                              hipStream_t stream) {
  const float* query = (const float*)d_in[0];
  const float* key_  = (const float*)d_in[1];
  const float* value = (const float*)d_in[2];
  const float* Wq = (const float*)d_in[3];
  const float* bq = (const float*)d_in[4];
  const float* Wk = (const float*)d_in[5];
  const float* bk = (const float*)d_in[6];
  const float* Wv = (const float*)d_in[7];
  const float* bv = (const float*)d_in[8];
  const float* Wo = (const float*)d_in[9];
  const float* bo = (const float*)d_in[10];
  float* out = (float*)d_out;

  const size_t NE = (size_t)BATCH * SEQ * HIDDEN;  // 4,194,304 elems
  _Float16* Qp = (_Float16*)d_ws;                  // [B,H,S,Dh]
  _Float16* Kp = Qp + NE;
  _Float16* Vp = Kp + NE;
  _Float16* Og = Vp + NE;                          // [B,S,HIDDEN]
  // total workspace: 4 * 8MB = 32MB (f16)

  dim3 ggrid(HIDDEN / 128, (BATCH * SEQ) / 128);   // (8, 32)
  gemm128<0><<<ggrid, 256, 0, stream>>>(query, nullptr, Wq, bq, Qp, nullptr);
  gemm128<0><<<ggrid, 256, 0, stream>>>(key_,  nullptr, Wk, bk, Kp, nullptr);
  gemm128<0><<<ggrid, 256, 0, stream>>>(value, nullptr, Wv, bv, Vp, nullptr);

  attn64<<<dim3(SEQ / 128, BATCH * HEADS), 128, 0, stream>>>(Qp, Kp, Vp, Og);

  gemm128<1><<<ggrid, 256, 0, stream>>>(nullptr, Og, Wo, bo, nullptr, out);
}
